// SinkhornAttention_48533130445201
// MI455X (gfx1250) — matrix operand order repaired and checked by reading, not compiled, not run on hardware
//
#include <hip/hip_runtime.h>
#include <hip/hip_bf16.h>

#define NBUCKETS 32
#define BSZ      128
#define DHDIM    64
#define TTOK     4096
#define BDIM     8192              // BSZ*DHDIM, elements per bucket panel
#define SCL      0.03125f          // 1024^-0.5
#define INVTEMP  (1.0f/0.75f)

typedef _Float16 v16h __attribute__((ext_vector_type(16)));
typedef _Float16 v8h  __attribute__((ext_vector_type(8)));
typedef float    v8f  __attribute__((ext_vector_type(8)));
typedef float    v4f  __attribute__((ext_vector_type(4)));
typedef unsigned int u32x4 __attribute__((ext_vector_type(4)));
typedef int          i32x4 __attribute__((ext_vector_type(4)));
typedef int          i32x8 __attribute__((ext_vector_type(8)));

__device__ __forceinline__ v16h cat8(v8h a, v8h b) {
  return __builtin_shufflevector(a, b, 0,1,2,3,4,5,6,7,8,9,10,11,12,13,14,15);
}

// A-fragment (16x32 f16): lane m: row m, K = e0+{0..7,16..23}; lane m+16: K = e0+{8..15,24..31}
__device__ __forceinline__ v16h ldsA(const _Float16* base, int stride, int e0, int lane) {
  int m = lane & 15;
  int koff = e0 + ((lane & 16) ? 8 : 0);
  const _Float16* p = base + m * stride + koff;
  v8h a = *reinterpret_cast<const v8h*>(p);
  v8h b = *reinterpret_cast<const v8h*>(p + 16);
  return cat8(a, b);
}

// B-fragment (32x16 f16): lane n: column n0+n, K = k0+{0..7,16..23}; lane n+16: K = k0+{8..15,24..31}
__device__ __forceinline__ v16h ldsB(const _Float16* base, int stride, int n0, int k0, int lane) {
  int n = lane & 15;
  int koff = k0 + ((lane & 16) ? 8 : 0);
  const _Float16* p = base + (n0 + n) * stride + koff;
  v8h a = *reinterpret_cast<const v8h*>(p);
  v8h b = *reinterpret_cast<const v8h*>(p + 16);
  return cat8(a, b);
}

// Q A-fragment from global f32 row (64 contiguous floats), cvt f32->f16
__device__ __forceinline__ v16h gldQ(const float* qrow, int e0, int lane) {
  int base = e0 + ((lane & 16) ? 8 : 0);
  const float* p = qrow + base;
  v4f f0 = *reinterpret_cast<const v4f*>(p);
  v4f f1 = *reinterpret_cast<const v4f*>(p + 4);
  v4f f2 = *reinterpret_cast<const v4f*>(p + 16);
  v4f f3 = *reinterpret_cast<const v4f*>(p + 20);
  v16h r;
#pragma unroll
  for (int i = 0; i < 4; ++i) {
    r[i]      = (_Float16)f0[i];
    r[4 + i]  = (_Float16)f1[i];
    r[8 + i]  = (_Float16)f2[i];
    r[12 + i] = (_Float16)f3[i];
  }
  return r;
}

// TDM: 2-D tile (tile_d0 x tile_d1 elements of 2B) global -> LDS, with LDS row padding.
// D# packing per CDNA5 ISA 8.3/8.4 (group0: count/lds/global/type, group1: dims+pad).
__device__ __forceinline__ void tdm_copy2d_f16(unsigned lds_off, const void* gptr,
                                               unsigned tile_d0, unsigned tile_d1,
                                               unsigned stride0,
                                               unsigned pad_int_code, unsigned pad_amt_code,
                                               unsigned pad_en) {
  unsigned long long ga = (unsigned long long)(size_t)gptr;
  u32x4 g0;
  g0[0] = 1u;                                          // count=1, user mode
  g0[1] = lds_off;                                     // lds_addr [63:32]
  g0[2] = (unsigned)(ga & 0xFFFFFFFFull);              // global_addr lo
  g0[3] = (unsigned)((ga >> 32) & 0x01FFFFFFull) | 0x80000000u;  // addr hi | type=2
  unsigned d0 = (1u << 16)                             // data_size = 2 bytes
              | (pad_en << 20)
              | (pad_int_code << 22)
              | (pad_amt_code << 25);
  unsigned long long td0 = tile_d0, td1 = tile_d1, st0 = stride0;
  unsigned long long w0 = (unsigned long long)d0 | ((td0 & 0xFFFFull) << 48);
  unsigned long long w1 = (td0 >> 16) | (td1 << 16) | ((td0 & 0xFFFFull) << 48); // tensor_dim0hi|tensor_dim1|tile_dim0
  unsigned long long w2 = (td1 & 0xFFFFull) | ((st0 & 0xFFFFFFFFull) << 32);     // tile_dim1|tile_dim2=0|stride0 lo
  unsigned long long w3 = (st0 >> 32);                                           // stride0 hi | stride1=0
  i32x8 g1;
  g1[0] = (int)(unsigned)w0; g1[1] = (int)(unsigned)(w0 >> 32);
  g1[2] = (int)(unsigned)w1; g1[3] = (int)(unsigned)(w1 >> 32);
  g1[4] = (int)(unsigned)w2; g1[5] = (int)(unsigned)(w2 >> 32);
  g1[6] = (int)(unsigned)w3; g1[7] = (int)(unsigned)(w3 >> 32);
  i32x4 z4 = {0, 0, 0, 0};
#if defined(__clang_major__) && (__clang_major__ >= 23)
  i32x8 z8 = {0, 0, 0, 0, 0, 0, 0, 0};
  __builtin_amdgcn_tensor_load_to_lds(g0, g1, z4, z4, z8, 0);
#else
  __builtin_amdgcn_tensor_load_to_lds(g0, g1, z4, z4, 0);
#endif
}

// ---------------- Phase 1: routing matrix R (per bh) -----------------------
__global__ __launch_bounds__(256)
void sinkhorn_phase1(const float* __restrict__ k, const float* __restrict__ sort_w,
                     const float* __restrict__ gumbel, float* __restrict__ Rout) {
  __shared__ float bkr[NBUCKETS][DHDIM];
  __shared__ float M[NBUCKETS][NBUCKETS + 1];
  const int bh  = blockIdx.x;
  const int h   = bh & 15;
  const int tid = threadIdx.x;
  const size_t kb = (size_t)bh * TTOK * DHDIM;

  for (int it = 0; it < 8; ++it) {
    int e = tid + it * 256;
    int bu = e >> 6, d = e & 63;
    float acc = 0.f;
    for (int t = 0; t < BSZ; ++t)
      acc += k[kb + ((size_t)bu * BSZ + t) * DHDIM + d];
    bkr[bu][d] = acc;
  }
  __syncthreads();

  const float* sw = sort_w + (size_t)h * DHDIM * NBUCKETS;
  for (int it = 0; it < 4; ++it) {
    int e = tid + it * 256;
    int uu = e >> 5, vv = e & 31;
    float acc = 0.f;
    for (int d = 0; d < DHDIM; ++d)
      acc += bkr[uu][d] * sw[d * NBUCKETS + vv];
    acc = fmaxf(acc, 0.f);
    float r = __logf(acc + 1e-6f);
    r = (r + gumbel[(size_t)bh * 1024 + uu * 32 + vv]) * INVTEMP;
    M[uu][vv] = r;
  }
  __syncthreads();

  for (int itr = 0; itr < 7; ++itr) {
    if (tid < 32) {
      float m = -3.0e38f;
      for (int j = 0; j < 32; ++j) m = fmaxf(m, M[tid][j]);
      float s = 0.f;
      for (int j = 0; j < 32; ++j) s += __expf(M[tid][j] - m);
      float lse = m + __logf(s);
      for (int j = 0; j < 32; ++j) M[tid][j] -= lse;
    }
    __syncthreads();
    if (tid < 32) {
      float m = -3.0e38f;
      for (int i = 0; i < 32; ++i) m = fmaxf(m, M[i][tid]);
      float s = 0.f;
      for (int i = 0; i < 32; ++i) s += __expf(M[i][tid] - m);
      float lse = m + __logf(s);
      for (int i = 0; i < 32; ++i) M[i][tid] -= lse;
    }
    __syncthreads();
  }

  for (int it = 0; it < 4; ++it) {
    int e = tid + it * 256;
    int uu = e >> 5, vv = e & 31;
    float val = (vv < uu) ? __expf(M[uu][vv]) : 0.f;
    Rout[(size_t)bh * 1024 + uu * 32 + vv] = val;
  }
}

// ---------------- Phase 2: bucket permute, once per bh ---------------------
// kre[u] = sum_{v<u} R[u][v]*kb[v]  (f16), kcv = f16(kb); same for v.
__global__ __launch_bounds__(256)
void sinkhorn_permute(const float* __restrict__ k, const float* __restrict__ v,
                      const float* __restrict__ R,
                      _Float16* __restrict__ kre, _Float16* __restrict__ kcv,
                      _Float16* __restrict__ vre, _Float16* __restrict__ vcv) {
  __shared__ float sRm[NBUCKETS][NBUCKETS];
  const int bh    = blockIdx.x;
  const int chunk = blockIdx.y;
  const int tid   = threadIdx.x;

  for (int i = 0; i < 4; ++i) {
    int e = tid + i * 256;
    sRm[e >> 5][e & 31] = R[(size_t)bh * 1024 + e];
  }
  __syncthreads();

  const size_t base = (size_t)bh * TTOK * DHDIM;
  for (int i = 0; i < 8; ++i) {
    const int td = chunk * 2048 + i * 256 + tid;
    float val[NBUCKETS];
    // ---- K ----
#pragma unroll
    for (int vb = 0; vb < NBUCKETS; ++vb) {
      val[vb] = k[base + (size_t)vb * BDIM + td];
      kcv[base + (size_t)vb * BDIM + td] = (_Float16)val[vb];
    }
#pragma unroll
    for (int u = 0; u < NBUCKETS; ++u) {
      float acc = 0.f;
#pragma unroll
      for (int vb = 0; vb < u; ++vb) acc += sRm[u][vb] * val[vb];
      kre[base + (size_t)u * BDIM + td] = (_Float16)acc;
    }
    // ---- V ----
#pragma unroll
    for (int vb = 0; vb < NBUCKETS; ++vb) {
      val[vb] = v[base + (size_t)vb * BDIM + td];
      vcv[base + (size_t)vb * BDIM + td] = (_Float16)val[vb];
    }
#pragma unroll
    for (int u = 0; u < NBUCKETS; ++u) {
      float acc = 0.f;
#pragma unroll
      for (int vb = 0; vb < u; ++vb) acc += sRm[u][vb] * val[vb];
      vre[base + (size_t)u * BDIM + td] = (_Float16)acc;
    }
  }
}

// ---------------- Phase 3: attention per (bh, bucket) ----------------------
__global__ __launch_bounds__(256)
void sinkhorn_attn(const float* __restrict__ q,
                   const _Float16* __restrict__ kre, const _Float16* __restrict__ kcv,
                   const _Float16* __restrict__ vre, const _Float16* __restrict__ vcv,
                   float* __restrict__ out) {
  __shared__ __align__(16) _Float16 sK[BSZ][72];      // key half-panel, row stride 144B
  __shared__ __align__(16) _Float16 sVT[DHDIM][264];  // bv transposed
  __shared__ __align__(16) _Float16 sP[8][16][40];    // per-wave P bounce

  const int u    = blockIdx.x;
  const int bh   = blockIdx.y;
  const int tid  = threadIdx.x;
  const int lane = tid & 31;
  const int wave = tid >> 5;

  const size_t pan = ((size_t)bh * NBUCKETS + u) * BDIM;  // panel offset (elements)

  // Stage bv^T (transpose through LDS): j<128 from vre, j>=128 from vcv
  for (int it = 0; it < 32; ++it) {
    int e = tid + it * 256;           // 0..8191
    int d2 = (e & 31) * 2;
    int j  = e >> 5;
    const _Float16* src = (j < BSZ) ? (vre + pan + (size_t)j * DHDIM + d2)
                                    : (vcv + pan + (size_t)(j - BSZ) * DHDIM + d2);
    unsigned w = *reinterpret_cast<const unsigned*>(src);
    union { unsigned u32; _Float16 h[2]; } cv; cv.u32 = w;
    sVT[d2][j]     = cv.h[0];
    sVT[d2 + 1][j] = cv.h[1];
  }

  // Q A-fragments (wave's 16 rows)
  const float* qrow = q + (size_t)bh * TTOK * DHDIM +
                      ((size_t)u * BSZ + wave * 16 + (lane & 15)) * DHDIM;
  v16h QA0 = gldQ(qrow, 0, lane);
  v16h QA1 = gldQ(qrow, 32, lane);

  // GEMM1: S = Q @ BK^T, key halves staged by the Tensor Data Mover
  v8f S[16];
  const unsigned sKoff = (unsigned)(size_t)(void*)&sK[0][0];
#pragma unroll
  for (int half = 0; half < 2; ++half) {
    __syncthreads();  // prior sK readers (and sVT fill) done before TDM overwrite
    if (wave == 0) {
      const _Float16* src = (half == 0) ? (kre + pan) : (kcv + pan);
      // 128 rows x 64 f16; LDS pad 4 DWORDs after every 32 DWORDs -> stride 144B
      tdm_copy2d_f16(sKoff, src, /*tile_d0=*/64, /*tile_d1=*/128, /*stride0=*/64,
                     /*pad_interval(32dw)=*/4, /*pad_amount(4dw)=*/3, /*pad_en=*/1);
      __builtin_amdgcn_s_wait_tensorcnt(0);
    }
    __syncthreads();
#pragma unroll
    for (int jt = 0; jt < 8; ++jt) {
      v8f c = {};
      v16h B0 = ldsB(&sK[0][0], 72, jt * 16, 0,  lane);
      v16h B1 = ldsB(&sK[0][0], 72, jt * 16, 32, lane);
      c = __builtin_amdgcn_wmma_f32_16x16x32_f16(false, QA0, false, B0, (short)0, c, false, false);
      c = __builtin_amdgcn_wmma_f32_16x16x32_f16(false, QA1, false, B1, (short)0, c, false, false);
      S[half * 8 + jt] = c;
    }
  }

  // Softmax over 256 keys (row in VGPR index, N across 16-lane halves)
  float rinv[8];
#pragma unroll
  for (int g = 0; g < 8; ++g) {
    float m = -3.0e38f;
#pragma unroll
    for (int t = 0; t < 16; ++t) { S[t][g] *= SCL; m = fmaxf(m, S[t][g]); }
#pragma unroll
    for (int o = 1; o < 16; o <<= 1) m = fmaxf(m, __shfl_xor(m, o, 16));
    float s = 0.f;
#pragma unroll
    for (int t = 0; t < 16; ++t) { float e_ = __expf(S[t][g] - m); S[t][g] = e_; s += e_; }
#pragma unroll
    for (int o = 1; o < 16; o <<= 1) s += __shfl_xor(s, o, 16);
    rinv[g] = 1.0f / s;
  }

  // GEMM2: O = P @ BV via per-wave LDS bounce (C-layout -> A-layout)
  v8f O0 = {}, O1 = {}, O2 = {}, O3 = {};
  _Float16* myP = &sP[wave][0][0];
  const int rb = (lane & 16) ? 8 : 0;
#pragma unroll
  for (int jc = 0; jc < 8; ++jc) {
#pragma unroll
    for (int tt = 0; tt < 2; ++tt) {
      const int t = jc * 2 + tt;
      const int col = tt * 16 + (lane & 15);
#pragma unroll
      for (int g = 0; g < 8; ++g)
        myP[(rb + g) * 40 + col] = (_Float16)(S[t][g] * rinv[g]);
    }
    asm volatile("s_wait_dscnt 0" ::: "memory");  // wave-private write->read ordering
    v16h A = ldsA(myP, 40, 0, lane);
    {
      v16h B = ldsB(&sVT[0][0], 264, 0,  jc * 32, lane);
      O0 = __builtin_amdgcn_wmma_f32_16x16x32_f16(false, A, false, B, (short)0, O0, false, false);
    }
    {
      v16h B = ldsB(&sVT[0][0], 264, 16, jc * 32, lane);
      O1 = __builtin_amdgcn_wmma_f32_16x16x32_f16(false, A, false, B, (short)0, O1, false, false);
    }
    {
      v16h B = ldsB(&sVT[0][0], 264, 32, jc * 32, lane);
      O2 = __builtin_amdgcn_wmma_f32_16x16x32_f16(false, A, false, B, (short)0, O2, false, false);
    }
    {
      v16h B = ldsB(&sVT[0][0], 264, 48, jc * 32, lane);
      O3 = __builtin_amdgcn_wmma_f32_16x16x32_f16(false, A, false, B, (short)0, O3, false, false);
    }
  }

  // Epilogue
  float* orow = out + (size_t)bh * TTOK * DHDIM + ((size_t)u * BSZ + wave * 16) * DHDIM;
  const int n = lane & 15;
#pragma unroll
  for (int g = 0; g < 8; ++g) {
    orow[(rb + g) * DHDIM + 0  + n] = O0[g];
    orow[(rb + g) * DHDIM + 16 + n] = O1[g];
    orow[(rb + g) * DHDIM + 32 + n] = O2[g];
    orow[(rb + g) * DHDIM + 48 + n] = O3[g];
  }
}

extern "C" void kernel_launch(void* const* d_in, const int* in_sizes, int n_in,
                              void* d_out, int out_size, void* d_ws, size_t ws_size,
                              hipStream_t stream) {
  (void)in_sizes; (void)n_in; (void)out_size; (void)ws_size;
  const float* q  = (const float*)d_in[0];
  const float* k  = (const float*)d_in[1];
  const float* v  = (const float*)d_in[2];
  const float* sw = (const float*)d_in[3];
  const float* gn = (const float*)d_in[4];

  // workspace layout
  const size_t PANEL = (size_t)64 * NBUCKETS * BDIM;   // 16,777,216 f16 elements
  char* ws = (char*)d_ws;
  float*    R   = (float*)ws;                          // 256 KB
  _Float16* kre = (_Float16*)(ws + (1u << 18));
  _Float16* kcv = kre + PANEL;
  _Float16* vre = kcv + PANEL;
  _Float16* vcv = vre + PANEL;

  sinkhorn_phase1 <<<dim3(64),      dim3(256), 0, stream>>>(k, sw, gn, R);
  sinkhorn_permute<<<dim3(64, 4),   dim3(256), 0, stream>>>(k, v, R, kre, kcv, vre, vcv);
  sinkhorn_attn   <<<dim3(32, 64),  dim3(256), 0, stream>>>(q, kre, kcv, vre, vcv, (float*)d_out);
}
